// EnnS2SEncoder_17051020165394
// MI455X (gfx1250) — matrix operand order repaired and checked by reading, not compiled, not run on hardware
//
#include <hip/hip_runtime.h>
#include <hip/hip_bf16.h>
#include <math.h>

typedef __attribute__((ext_vector_type(16))) _Float16 v16h;
typedef __attribute__((ext_vector_type(8)))  _Float16 v8h;
typedef __attribute__((ext_vector_type(8)))  float    v8f;
typedef __attribute__((ext_vector_type(4)))  float    v4f;
typedef __attribute__((ext_vector_type(4)))  int      v4i;

#define N_NODESC 50000
#define N_EDGESC 200000
#define NFEAT 16
#define DIM 32
#define NGRAPH 1000
#define N_TILES (N_EDGESC / 16)      /* 12500 */
#define NSLICES 129                  /* 128 rank-1 (i,k-chunk) slices + 1 bias slice */
#define BFRAG_HALVES (NSLICES * 2 * 32 * 16)  /* 132096 halves = 258 KB */

#if defined(__gfx1250__) && __has_builtin(__builtin_amdgcn_global_load_async_to_lds_b128)
#define USE_ASYNC_LDS 1
#endif

/* ------------------------------------------------------------------ */
/* generic fill                                                        */
__global__ void fill_f32(float* __restrict__ p, long n, float v) {
  long i = (long)blockIdx.x * blockDim.x + threadIdx.x;
  if (i < n) p[i] = v;
}

/* ------------------------------------------------------------------ */
/* lin0: feat = relu(x @ W.T + b)                                      */
__global__ void lin0_kernel(const float* __restrict__ x, const float* __restrict__ W,
                            const float* __restrict__ b, float* __restrict__ feat) {
  __shared__ float Ws[DIM * NFEAT];
  __shared__ float bs[DIM];
  for (int i = threadIdx.x; i < DIM * NFEAT; i += blockDim.x) Ws[i] = W[i];
  for (int i = threadIdx.x; i < DIM; i += blockDim.x) bs[i] = b[i];
  __syncthreads();
  int n = blockIdx.x * blockDim.x + threadIdx.x;
  if (n >= N_NODESC) return;
  float xr[NFEAT];
#pragma unroll
  for (int k = 0; k < NFEAT; k++) xr[k] = x[(size_t)n * NFEAT + k];
  for (int g = 0; g < DIM; g++) {
    float acc = bs[g];
#pragma unroll
    for (int k = 0; k < NFEAT; k++) acc += Ws[g * NFEAT + k] * xr[k];
    feat[(size_t)n * DIM + g] = fmaxf(acc, 0.f);
  }
}

/* ------------------------------------------------------------------ */
/* edge MLP layer 1: h_mid = relu(edge_attr @ W1.T + b1), stored f16   */
__global__ void edge_mlp_kernel(const float* __restrict__ ea, const float* __restrict__ W1,
                                const float* __restrict__ b1, _Float16* __restrict__ hmid) {
  __shared__ float Ws[128 * 5];
  __shared__ float bs[128];
  for (int i = threadIdx.x; i < 128 * 5; i += blockDim.x) Ws[i] = W1[i];
  for (int i = threadIdx.x; i < 128; i += blockDim.x) bs[i] = b1[i];
  __syncthreads();
  int e = blockIdx.x * blockDim.x + threadIdx.x;
  if (e >= N_EDGESC) return;
  float a0 = ea[e * 5 + 0], a1 = ea[e * 5 + 1], a2 = ea[e * 5 + 2];
  float a3 = ea[e * 5 + 3], a4 = ea[e * 5 + 4];
  for (int g0 = 0; g0 < 128; g0 += 8) {
    v8h pack;
#pragma unroll
    for (int j = 0; j < 8; j++) {
      int g = g0 + j;
      float v = bs[g] + Ws[g * 5 + 0] * a0 + Ws[g * 5 + 1] * a1 + Ws[g * 5 + 2] * a2 +
                Ws[g * 5 + 3] * a3 + Ws[g * 5 + 4] * a4;
      pack[j] = (_Float16)fmaxf(v, 0.f);
    }
    *(v8h*)(hmid + (size_t)e * 128 + g0) = pack;
  }
}

/* ------------------------------------------------------------------ */
/* Pack mlp_W2 (+ b2 as extra K-slice) into per-lane WMMA B fragments. */
/* Fragment (s, nb), lane l, elem t:  K = 8*(l>>4)+(t&7)+16*(t>>3), N = (l&15). */
/* s<128: i=s>>2, c=s&3, k=32*c+K, value = W2[(i*32 + nb*16+N)*128+k]  */
/* s==128 (bias slice, A = x itself): value = b2[K*32 + nb*16 + N]     */
__global__ void bpack_kernel(const float* __restrict__ W2, const float* __restrict__ b2,
                             _Float16* __restrict__ Bpk) {
  int idx = blockIdx.x * blockDim.x + threadIdx.x;
  if (idx >= BFRAG_HALVES) return;
  int t = idx & 15;
  int l = (idx >> 4) & 31;
  int nb = (idx >> 9) & 1;
  int s = idx >> 10;
  int K = 8 * (l >> 4) + (t & 7) + 16 * (t >> 3);
  int n = nb * 16 + (l & 15);
  float v;
  if (s < 128) {
    int i = s >> 2, c = s & 3;
    int k = 32 * c + K;
    v = W2[(size_t)(i * 32 + n) * 128 + k];
  } else {
    v = b2[K * 32 + n];
  }
  Bpk[idx] = (_Float16)v;
}

/* ------------------------------------------------------------------ */
/* degree = segment_sum(ones, dst)                                     */
__global__ void deg_kernel(const int* __restrict__ ei, float* __restrict__ deg) {
  int e = blockIdx.x * blockDim.x + threadIdx.x;
  if (e >= N_EDGESC) return;
  atomicAdd(deg + ei[N_EDGESC + e], 1.0f);
}

/* ------------------------------------------------------------------ */
/* Fused NNConv: per 16-edge tile, msg = (x_src ⊗ h_mid) @ Bpk via     */
/* v_wmma_f32_16x16x32_f16, then atomic scatter-add into agg[dst].     */
__device__ __forceinline__ v8f wmma16(v16h a, v16h b, v8f c) {
  return __builtin_amdgcn_wmma_f32_16x16x32_f16(false, a, false, b, (short)0, c, false, false);
}

__device__ __forceinline__ v16h load_bfrag(const _Float16* Blds, int s, int nb, int lane) {
  /* 32B contiguous per lane -> two ds_load_b128, no element shuffles */
  return *(const v16h*)(Blds + ((size_t)((s * 2 + nb) * 32 + lane) << 4));
}

__global__ void __launch_bounds__(256, 1)
nnconv_wmma_kernel(const float* __restrict__ feat, const _Float16* __restrict__ hmid,
                   const _Float16* __restrict__ Bpk, const int* __restrict__ ei,
                   float* __restrict__ agg) {
  extern __shared__ _Float16 Blds[];  /* 132096 halves = 258 KB (fragment-ordered) */

  /* ---- stage B fragments into LDS ---- */
#ifdef USE_ASYNC_LDS
  {
    /* CDNA5 async bulk copy: global -> LDS without VGPR round trip      */
    for (int i = threadIdx.x * 8; i < BFRAG_HALVES; i += blockDim.x * 8) {
      __builtin_amdgcn_global_load_async_to_lds_b128((v4i*)(Bpk + i), (v4i*)(Blds + i), 0, 0);
    }
#if __has_builtin(__builtin_amdgcn_s_wait_asynccnt)
    __builtin_amdgcn_s_wait_asynccnt(0);
#else
    asm volatile("s_wait_asynccnt 0" ::: "memory");
#endif
  }
#else
  {
    const v8h* g = (const v8h*)Bpk;
    v8h* s = (v8h*)Blds;
    for (int i = threadIdx.x; i < BFRAG_HALVES / 8; i += blockDim.x) s[i] = g[i];
  }
#endif
  __syncthreads();

  const int lane = threadIdx.x & 31;
  const int eL = lane & 15;          /* edge within tile (row M for A)            */
  const int half = lane >> 4;        /* lane-half selects K octets per ISA layout */
  const int col = lane & 15;         /* C/D column                                */
  const int waveId = (blockIdx.x * blockDim.x + threadIdx.x) >> 5;
  const int nWaves = (gridDim.x * blockDim.x) >> 5;

  for (int tile = waveId; tile < N_TILES; tile += nWaves) {
    const int e = tile * 16 + eL;
    const int srcN = ei[e];
    const int dstN = ei[N_EDGESC + e];

    /* prefetch next tile's h_mid row (global_prefetch_b8) */
    {
      int tn = tile + nWaves;
      if (tn < N_TILES) __builtin_prefetch(hmid + (size_t)(tn * 16 + eL) * 128, 0, 1);
    }

    /* source-node feature row (32 f32) */
    const float* xr = feat + (size_t)srcN * DIM;
    float xv[32];
#pragma unroll
    for (int q = 0; q < 8; q++) {
      v4f p = *(const v4f*)(xr + q * 4);
      xv[q * 4 + 0] = p[0]; xv[q * 4 + 1] = p[1];
      xv[q * 4 + 2] = p[2]; xv[q * 4 + 3] = p[3];
    }

    /* h_mid fragments: chunk c covers h[32c..32c+31]; this lane-half's octets */
    const _Float16* hr = hmid + (size_t)e * 128;
    v16h hf[4];
#pragma unroll
    for (int c = 0; c < 4; c++) {
      v8h lo = *(const v8h*)(hr + 32 * c + 8 * half);
      v8h hi = *(const v8h*)(hr + 32 * c + 16 + 8 * half);
      hf[c] = __builtin_shufflevector(lo, hi, 0, 1, 2, 3, 4, 5, 6, 7,
                                      8, 9, 10, 11, 12, 13, 14, 15);
    }

    /* bias-slice A fragment: x itself as f16, same octet layout */
    v16h ax;
    {
      v4f f0 = *(const v4f*)(xr + 8 * half);
      v4f f1 = *(const v4f*)(xr + 8 * half + 4);
      v4f f2 = *(const v4f*)(xr + 16 + 8 * half);
      v4f f3 = *(const v4f*)(xr + 16 + 8 * half + 4);
      v8f lo8 = __builtin_shufflevector(f0, f1, 0, 1, 2, 3, 4, 5, 6, 7);
      v8f hi8 = __builtin_shufflevector(f2, f3, 0, 1, 2, 3, 4, 5, 6, 7);
      v8h lo = __builtin_convertvector(lo8, v8h);
      v8h hi = __builtin_convertvector(hi8, v8h);
      ax = __builtin_shufflevector(lo, hi, 0, 1, 2, 3, 4, 5, 6, 7,
                                   8, 9, 10, 11, 12, 13, 14, 15);
    }

    v8f acc0 = {};
    v8f acc1 = {};

    /* K sweep: 128 rank-1 slices, A = x[:,i] * h_chunk built in registers */
#pragma unroll
    for (int i = 0; i < 32; i++) {
      const _Float16 xh = (_Float16)xv[i];
#pragma unroll
      for (int c = 0; c < 4; c++) {
        v16h a = hf[c] * xh;                      /* 8x v_pk_mul_f16 */
        const int s = i * 4 + c;
        v16h b0 = load_bfrag(Blds, s, 0, lane);
        acc0 = wmma16(a, b0, acc0);
        v16h b1 = load_bfrag(Blds, s, 1, lane);
        acc1 = wmma16(a, b1, acc1);
      }
    }
    /* bias slice (accounts for mlp_b2): msg += x_src @ reshape(b2,32,32) */
    {
      v16h b0 = load_bfrag(Blds, 128, 0, lane);
      acc0 = wmma16(ax, b0, acc0);
      v16h b1 = load_bfrag(Blds, 128, 1, lane);
      acc1 = wmma16(ax, b1, acc1);
    }

    /* scatter-add: C layout -> VGPR j holds row M=j+8*half, col=lane&15 */
#pragma unroll
    for (int j = 0; j < 8; j++) {
      int dstM = __shfl(dstN, j + 8 * half, 32);
      float* base = agg + (size_t)dstM * DIM;
      atomicAdd(base + col, acc0[j]);
      atomicAdd(base + 16 + col, acc1[j]);
    }
  }
}

/* ------------------------------------------------------------------ */
/* GRU cell: m = relu(agg/deg + conv_bias); h = GRU(m, h); in place    */
__global__ void gru_kernel(const float* __restrict__ agg, const float* __restrict__ deg,
                           const float* __restrict__ cbias, const float* __restrict__ wih,
                           const float* __restrict__ whh, const float* __restrict__ bih,
                           const float* __restrict__ bhh, float* __restrict__ feat) {
  __shared__ float sWih[96 * DIM];
  __shared__ float sWhh[96 * DIM];
  __shared__ float sBih[96], sBhh[96], sCb[DIM];
  for (int i = threadIdx.x; i < 96 * DIM; i += blockDim.x) { sWih[i] = wih[i]; sWhh[i] = whh[i]; }
  for (int i = threadIdx.x; i < 96; i += blockDim.x) { sBih[i] = bih[i]; sBhh[i] = bhh[i]; }
  for (int i = threadIdx.x; i < DIM; i += blockDim.x) sCb[i] = cbias[i];
  __syncthreads();
  int n = blockIdx.x * blockDim.x + threadIdx.x;
  if (n >= N_NODESC) return;
  float invd = 1.0f / fmaxf(deg[n], 1.0f);
  float m[DIM], h[DIM];
#pragma unroll
  for (int k = 0; k < DIM; k++) {
    m[k] = fmaxf(agg[(size_t)n * DIM + k] * invd + sCb[k], 0.f);
    h[k] = feat[(size_t)n * DIM + k];
  }
  for (int g = 0; g < DIM; g++) {
    float ir = sBih[g], iz = sBih[32 + g], in_ = sBih[64 + g];
    float hr = sBhh[g], hz = sBhh[32 + g], hn = sBhh[64 + g];
#pragma unroll
    for (int k = 0; k < DIM; k++) {
      ir += sWih[g * DIM + k] * m[k];
      iz += sWih[(32 + g) * DIM + k] * m[k];
      in_ += sWih[(64 + g) * DIM + k] * m[k];
      hr += sWhh[g * DIM + k] * h[k];
      hz += sWhh[(32 + g) * DIM + k] * h[k];
      hn += sWhh[(64 + g) * DIM + k] * h[k];
    }
    float r = 1.f / (1.f + __expf(-(ir + hr)));
    float z = 1.f / (1.f + __expf(-(iz + hz)));
    float nn = tanhf(in_ + r * hn);
    float hg = feat[(size_t)n * DIM + g];  /* old h[g] (not yet overwritten) */
    feat[(size_t)n * DIM + g] = (1.f - z) * nn + z * hg;
  }
}

/* ------------------------------------------------------------------ */
/* Set2Set LSTM step: one block per graph, 128 threads = gate rows     */
__global__ void lstm_kernel(const float* __restrict__ qstar, const float* __restrict__ wih,
                            const float* __restrict__ whh, const float* __restrict__ bih,
                            const float* __restrict__ bhh, float* __restrict__ hs,
                            float* __restrict__ cs) {
  int g = threadIdx.x;
  int graph = blockIdx.x;
  __shared__ float gates[128];
  const float* q = qstar + (size_t)graph * 64;
  const float* h = hs + (size_t)graph * 32;
  float acc = bih[g] + bhh[g];
  for (int k = 0; k < 64; k++) acc += wih[g * 64 + k] * q[k];
  for (int k = 0; k < 32; k++) acc += whh[g * 32 + k] * h[k];
  gates[g] = acc;
  __syncthreads();
  if (g < 32) {
    float ii = 1.f / (1.f + __expf(-gates[g]));
    float ff = 1.f / (1.f + __expf(-gates[32 + g]));
    float gg = tanhf(gates[64 + g]);
    float oo = 1.f / (1.f + __expf(-gates[96 + g]));
    float c = ff * cs[(size_t)graph * 32 + g] + ii * gg;
    cs[(size_t)graph * 32 + g] = c;
    hs[(size_t)graph * 32 + g] = oo * tanhf(c);
  }
}

/* ------------------------------------------------------------------ */
/* segment softmax attention + weighted pooling                        */
__device__ __forceinline__ void atomicMaxF(float* addr, float v) {
  if (v >= 0.f) atomicMax((int*)addr, __float_as_int(v));
  else atomicMin((unsigned int*)addr, __float_as_uint(v));
}

__global__ void attn_e_kernel(const float* __restrict__ feat, const float* __restrict__ hs,
                              const int* __restrict__ batch, float* __restrict__ ebuf,
                              float* __restrict__ mseg) {
  int n = blockIdx.x * blockDim.x + threadIdx.x;
  if (n >= N_NODESC) return;
  int b = batch[n];
  const float* f = feat + (size_t)n * DIM;
  const float* q = hs + (size_t)b * DIM;
  float e = 0.f;
#pragma unroll
  for (int k = 0; k < DIM; k++) e += f[k] * q[k];
  ebuf[n] = e;
  atomicMaxF(mseg + b, e);
}

__global__ void attn_exp_kernel(float* __restrict__ ebuf, const float* __restrict__ mseg,
                                float* __restrict__ sseg, const int* __restrict__ batch) {
  int n = blockIdx.x * blockDim.x + threadIdx.x;
  if (n >= N_NODESC) return;
  int b = batch[n];
  float a = __expf(ebuf[n] - mseg[b]);
  ebuf[n] = a;
  atomicAdd(sseg + b, a);
}

__global__ void attn_r_kernel(const float* __restrict__ feat, const float* __restrict__ ebuf,
                              const float* __restrict__ sseg, const int* __restrict__ batch,
                              float* __restrict__ rseg) {
  int t = blockIdx.x * blockDim.x + threadIdx.x;
  if (t >= N_NODESC * DIM) return;
  int n = t >> 5, d = t & 31;
  int b = batch[n];
  float wgt = ebuf[n] / sseg[b];
  atomicAdd(rseg + (size_t)b * DIM + d, wgt * feat[(size_t)n * DIM + d]);
}

__global__ void qstar_kernel(const float* __restrict__ hs, const float* __restrict__ rseg,
                             float* __restrict__ qstar) {
  int t = blockIdx.x * blockDim.x + threadIdx.x;
  if (t >= NGRAPH * DIM) return;
  int g = t >> 5, d = t & 31;
  qstar[(size_t)g * 64 + d] = hs[t];
  qstar[(size_t)g * 64 + 32 + d] = rseg[t];
}

/* ------------------------------------------------------------------ */
__global__ void output_kernel(const float* __restrict__ qstar, const float* __restrict__ feat,
                              float* __restrict__ out) {
  long i = (long)blockIdx.x * blockDim.x + threadIdx.x;
  const long nq = (long)NGRAPH * 64;
  const long nf = (long)N_NODESC * DIM;
  if (i < nq) out[i] = qstar[i];
  else if (i < nq + nf) out[i] = feat[i - nq];
}

/* ================================================================== */
extern "C" void kernel_launch(void* const* d_in, const int* in_sizes, int n_in,
                              void* d_out, int out_size, void* d_ws, size_t ws_size,
                              hipStream_t stream) {
  const float* x         = (const float*)d_in[0];
  const float* edge_attr = (const float*)d_in[1];
  const int*   edge_idx  = (const int*)d_in[2];
  const int*   batch     = (const int*)d_in[3];
  const float* lin0_W    = (const float*)d_in[4];
  const float* lin0_b    = (const float*)d_in[5];
  const float* mlp_W1    = (const float*)d_in[6];
  const float* mlp_b1    = (const float*)d_in[7];
  const float* mlp_W2    = (const float*)d_in[8];
  const float* mlp_b2    = (const float*)d_in[9];
  const float* conv_bias = (const float*)d_in[10];
  const float* gru_w_ih  = (const float*)d_in[11];
  const float* gru_w_hh  = (const float*)d_in[12];
  const float* gru_b_ih  = (const float*)d_in[13];
  const float* gru_b_hh  = (const float*)d_in[14];
  const float* lstm_w_ih = (const float*)d_in[15];
  const float* lstm_w_hh = (const float*)d_in[16];
  const float* lstm_b_ih = (const float*)d_in[17];
  const float* lstm_b_hh = (const float*)d_in[18];

  char* w = (char*)d_ws;
  auto alloc = [&](size_t bytes) -> char* {
    char* p = w;
    w += (bytes + 255) & ~(size_t)255;
    return p;
  };
  float* feat  = (float*)alloc((size_t)N_NODESC * DIM * 4);
  float* agg   = (float*)alloc((size_t)N_NODESC * DIM * 4);
  float* degv  = (float*)alloc((size_t)N_NODESC * 4);
  float* hsb   = (float*)alloc((size_t)NGRAPH * DIM * 4);
  float* csb   = (float*)alloc((size_t)NGRAPH * DIM * 4);
  float* qstar = (float*)alloc((size_t)NGRAPH * 64 * 4);
  float* ebuf  = (float*)alloc((size_t)N_NODESC * 4);
  float* mseg  = (float*)alloc((size_t)NGRAPH * 4);
  float* sseg  = (float*)alloc((size_t)NGRAPH * 4);
  float* rseg  = (float*)alloc((size_t)NGRAPH * DIM * 4);
  _Float16* hmid = (_Float16*)alloc((size_t)N_EDGESC * 128 * 2);
  _Float16* Bpk  = (_Float16*)alloc((size_t)BFRAG_HALVES * 2);

  auto grid = [](long n) { return (unsigned)((n + 255) / 256); };

  /* ---- setup (iteration-invariant) ---- */
  fill_f32<<<grid(N_NODESC), 256, 0, stream>>>(degv, N_NODESC, 0.f);
  deg_kernel<<<grid(N_EDGESC), 256, 0, stream>>>(edge_idx, degv);
  lin0_kernel<<<grid(N_NODESC), 256, 0, stream>>>(x, lin0_W, lin0_b, feat);
  edge_mlp_kernel<<<grid(N_EDGESC), 256, 0, stream>>>(edge_attr, mlp_W1, mlp_b1, hmid);
  bpack_kernel<<<grid(BFRAG_HALVES), 256, 0, stream>>>(mlp_W2, mlp_b2, Bpk);

  /* ---- 3 NNConv + GRU iterations ---- */
  const size_t convLds = (size_t)BFRAG_HALVES * sizeof(_Float16); /* 258 KB */
  for (int it = 0; it < 3; it++) {
    fill_f32<<<grid((long)N_NODESC * DIM), 256, 0, stream>>>(agg, (long)N_NODESC * DIM, 0.f);
    nnconv_wmma_kernel<<<256, 256, convLds, stream>>>(feat, hmid, Bpk, edge_idx, agg);
    gru_kernel<<<grid(N_NODESC), 256, 0, stream>>>(agg, degv, conv_bias, gru_w_ih, gru_w_hh,
                                                   gru_b_ih, gru_b_hh, feat);
  }

  /* ---- Set2Set pooling ---- */
  fill_f32<<<grid((long)NGRAPH * DIM), 256, 0, stream>>>(hsb, (long)NGRAPH * DIM, 0.f);
  fill_f32<<<grid((long)NGRAPH * DIM), 256, 0, stream>>>(csb, (long)NGRAPH * DIM, 0.f);
  fill_f32<<<grid((long)NGRAPH * 64), 256, 0, stream>>>(qstar, (long)NGRAPH * 64, 0.f);
  for (int t = 0; t < 3; t++) {
    fill_f32<<<grid(NGRAPH), 256, 0, stream>>>(mseg, NGRAPH, -HUGE_VALF);
    fill_f32<<<grid(NGRAPH), 256, 0, stream>>>(sseg, NGRAPH, 0.f);
    fill_f32<<<grid((long)NGRAPH * DIM), 256, 0, stream>>>(rseg, (long)NGRAPH * DIM, 0.f);
    lstm_kernel<<<NGRAPH, 128, 0, stream>>>(qstar, lstm_w_ih, lstm_w_hh, lstm_b_ih, lstm_b_hh,
                                            hsb, csb);
    attn_e_kernel<<<grid(N_NODESC), 256, 0, stream>>>(feat, hsb, batch, ebuf, mseg);
    attn_exp_kernel<<<grid(N_NODESC), 256, 0, stream>>>(ebuf, mseg, sseg, batch);
    attn_r_kernel<<<grid((long)N_NODESC * DIM), 256, 0, stream>>>(feat, ebuf, sseg, batch, rseg);
    qstar_kernel<<<grid((long)NGRAPH * DIM), 256, 0, stream>>>(hsb, rseg, qstar);
  }

  /* ---- outputs: [q_star (1000x64), feat (50000x32)] ---- */
  output_kernel<<<grid((long)NGRAPH * 64 + (long)N_NODESC * DIM), 256, 0, stream>>>(
      qstar, feat, (float*)d_out);
}
